// MolecularGraphNeuralNetwork_49417893707860
// MI455X (gfx1250) — compile-verified, matrix operand order checked by reading, not run on hardware
//
#include <hip/hip_runtime.h>
#include <hip/hip_bf16.h>
#include <stdint.h>

#define N_ATOMS 8192
#define DIM 64
#define N_LAYERS 3
#define N_MOL 256
#define KSPLIT 4
#define KCHUNK (N_ATOMS / KSPLIT)

typedef __attribute__((ext_vector_type(16))) __bf16 v16bf;
typedef __attribute__((ext_vector_type(8)))  float  v8f;

union Frag32B { v16bf v; uint4 q[2]; };

__device__ __forceinline__ unsigned short f32_to_bf16_rne(float f) {
  unsigned int u = __float_as_uint(f);
  u += 0x7FFFu + ((u >> 16) & 1u);
  return (unsigned short)(u >> 16);
}

// ---------------------------------------------------------------------------
// Convert adjacency f32 -> bf16 (8 elements / thread, b128 in/out)
// ---------------------------------------------------------------------------
__global__ __launch_bounds__(256)
void cvt_a_bf16(const float* __restrict__ A, unsigned short* __restrict__ Abf) {
  size_t i = (size_t)blockIdx.x * 256u + threadIdx.x;      // 8Mi threads, 8 elems each
  const float4* src = reinterpret_cast<const float4*>(A) + i * 2;
  float4 x = src[0];
  float4 y = src[1];
  uint4 o;
  o.x = (unsigned)f32_to_bf16_rne(x.x) | ((unsigned)f32_to_bf16_rne(x.y) << 16);
  o.y = (unsigned)f32_to_bf16_rne(x.z) | ((unsigned)f32_to_bf16_rne(x.w) << 16);
  o.z = (unsigned)f32_to_bf16_rne(y.x) | ((unsigned)f32_to_bf16_rne(y.y) << 16);
  o.w = (unsigned)f32_to_bf16_rne(y.z) | ((unsigned)f32_to_bf16_rne(y.w) << 16);
  reinterpret_cast<uint4*>(Abf)[i] = o;
}

// ---------------------------------------------------------------------------
// Embedding gather: V[n][:] = table[fp[n]][:]  (one float4 per thread)
// ---------------------------------------------------------------------------
__global__ __launch_bounds__(256)
void embed_gather(const int* __restrict__ fp, const float* __restrict__ tbl,
                  float* __restrict__ V) {
  int i = blockIdx.x * 256 + threadIdx.x;                  // 131072 float4s
  int n = i >> 4;                                          // DIM/4 = 16 float4 per row
  int c = i & 15;
  int f = fp[n];
  reinterpret_cast<float4*>(V)[i] =
      reinterpret_cast<const float4*>(tbl + (size_t)f * DIM)[c];
}

// ---------------------------------------------------------------------------
// h = relu(V @ W^T + b);  writes h (f32 row-major) and Ht (bf16, [DIM][N_ATOMS])
// Block: 256 threads = 4 atom rows x 64 features. W staged in LDS.
// ---------------------------------------------------------------------------
__global__ __launch_bounds__(256)
void linear_relu(const float* __restrict__ V, const float* __restrict__ W,
                 const float* __restrict__ bias, float* __restrict__ h,
                 unsigned short* __restrict__ Ht) {
  __shared__ float Ws[DIM * DIM];
  __shared__ float Vs[4 * DIM];
  int tid = threadIdx.x;
  int n0  = blockIdx.x * 4;
  for (int i = tid; i < DIM * DIM; i += 256) Ws[i] = W[i];
  Vs[tid] = V[(size_t)n0 * DIM + tid];
  __syncthreads();

  int nl = tid >> 6;
  int e  = tid & 63;
  const float* wr = Ws + e * DIM;
  const float* vr = Vs + nl * DIM;
  float acc = bias[e];
#pragma unroll
  for (int d = 0; d < DIM; ++d) acc = fmaf(vr[d], wr[d], acc);
  acc = acc > 0.0f ? acc : 0.0f;

  int n = n0 + nl;
  h[(size_t)n * DIM + e] = acc;
  Ht[(size_t)e * N_ATOMS + n] = f32_to_bf16_rne(acc);
}

// ---------------------------------------------------------------------------
// Message-passing GEMM: partial[p] = A_bf16[m-stripe, k-chunk] @ H
// via v_wmma_f32_16x16x32_bf16, software-pipelined with register double
// buffering so loads for step k+32 are in flight while WMMAs consume step k.
// One wave -> 16 rows x 64 cols (4 acc tiles); K split KSPLIT ways.
// ---------------------------------------------------------------------------
struct Frags {
  Frag32B a, b0, b1, b2, b3;
};

__device__ __forceinline__ void load_frags(Frags& f,
                                           const unsigned short* __restrict__ arow,
                                           const unsigned short* __restrict__ brow,
                                           int k) {
  const unsigned short* ap = arow + k;
  f.a.q[0]  = *reinterpret_cast<const uint4*>(ap);
  f.a.q[1]  = *reinterpret_cast<const uint4*>(ap + 16);
  const unsigned short* bp = brow + k;
  f.b0.q[0] = *reinterpret_cast<const uint4*>(bp);
  f.b0.q[1] = *reinterpret_cast<const uint4*>(bp + 16);
  f.b1.q[0] = *reinterpret_cast<const uint4*>(bp + 16 * N_ATOMS);
  f.b1.q[1] = *reinterpret_cast<const uint4*>(bp + 16 * N_ATOMS + 16);
  f.b2.q[0] = *reinterpret_cast<const uint4*>(bp + 32 * N_ATOMS);
  f.b2.q[1] = *reinterpret_cast<const uint4*>(bp + 32 * N_ATOMS + 16);
  f.b3.q[0] = *reinterpret_cast<const uint4*>(bp + 48 * N_ATOMS);
  f.b3.q[1] = *reinterpret_cast<const uint4*>(bp + 48 * N_ATOMS + 16);
}

__device__ __forceinline__ void mma4(const Frags& f, v8f& c0, v8f& c1, v8f& c2, v8f& c3) {
  c0 = __builtin_amdgcn_wmma_f32_16x16x32_bf16(false, f.a.v, false, f.b0.v, (short)0, c0, false, false);
  c1 = __builtin_amdgcn_wmma_f32_16x16x32_bf16(false, f.a.v, false, f.b1.v, (short)0, c1, false, false);
  c2 = __builtin_amdgcn_wmma_f32_16x16x32_bf16(false, f.a.v, false, f.b2.v, (short)0, c2, false, false);
  c3 = __builtin_amdgcn_wmma_f32_16x16x32_bf16(false, f.a.v, false, f.b3.v, (short)0, c3, false, false);
}

__global__ __launch_bounds__(128)
void gemm_mp(const unsigned short* __restrict__ Abf,
             const unsigned short* __restrict__ Ht,
             float* __restrict__ partial) {
  const int wave  = blockIdx.x * 4 + (threadIdx.x >> 5);
  const int mt    = wave >> 2;            // row tile 0..511
  const int p     = wave & (KSPLIT - 1);  // k-chunk
  const int lane  = threadIdx.x & 31;
  const int l16   = lane & 15;
  const int half8 = (lane >> 4) << 3;     // 0 or 8

  const int m0 = mt << 4;
  // fold per-lane half offset into the base pointers
  const unsigned short* arow = Abf + (size_t)(m0 + l16) * N_ATOMS + half8;
  const unsigned short* brow = Ht + (size_t)l16 * N_ATOMS + half8;

  v8f c0 = {}, c1 = {}, c2 = {}, c3 = {};

  int k = p * KCHUNK;
  Frags f0, f1;
  load_frags(f0, arow, brow, k);

#pragma unroll 1
  for (int it = 0; it < (KCHUNK / 64) - 1; ++it) {
    load_frags(f1, arow, brow, k + 32);     // in flight while f0 is consumed
    mma4(f0, c0, c1, c2, c3);
    load_frags(f0, arow, brow, k + 64);     // in flight while f1 is consumed
    mma4(f1, c0, c1, c2, c3);
    k += 64;
  }
  load_frags(f1, arow, brow, k + 32);
  mma4(f0, c0, c1, c2, c3);
  mma4(f1, c0, c1, c2, c3);

  // C/D layout: lane half selects M=0..7 vs M=8..15 (vgpr j), N = l16 (+e0)
  float* pr = partial + (size_t)p * N_ATOMS * DIM;
  int rbase = m0 + half8;
#pragma unroll
  for (int j = 0; j < 8; ++j) {
    float* dst = pr + (size_t)(rbase + j) * DIM + l16;
    dst[0]  = c0[j];
    dst[16] = c1[j];
    dst[32] = c2[j];
    dst[48] = c3[j];
  }
}

// ---------------------------------------------------------------------------
// V = h + sum_p partial[p]   (residual message passing combine), float4/thread
// ---------------------------------------------------------------------------
__global__ __launch_bounds__(256)
void combine(const float* __restrict__ h, const float* __restrict__ partial,
             float* __restrict__ V) {
  int i = blockIdx.x * 256 + threadIdx.x;   // 131072 float4s
  float4 s = reinterpret_cast<const float4*>(h)[i];
#pragma unroll
  for (int p = 0; p < KSPLIT; ++p) {
    float4 q = reinterpret_cast<const float4*>(partial + (size_t)p * N_ATOMS * DIM)[i];
    s.x += q.x; s.y += q.y; s.z += q.z; s.w += q.w;
  }
  reinterpret_cast<float4*>(V)[i] = s;
}

// ---------------------------------------------------------------------------
// Segment sum over sorted segment ids; one block per molecule, 64 lanes = dims
// ---------------------------------------------------------------------------
__device__ __forceinline__ int lower_bound(const int* __restrict__ seg, int val) {
  int lo = 0, hi = N_ATOMS;
  while (lo < hi) {
    int mid = (lo + hi) >> 1;
    if (seg[mid] < val) lo = mid + 1; else hi = mid;
  }
  return lo;
}

__global__ __launch_bounds__(64)
void segment_sum(const float* __restrict__ V, const int* __restrict__ seg,
                 float* __restrict__ out) {
  int m = blockIdx.x;
  int e = threadIdx.x;
  int start = lower_bound(seg, m);
  int end   = lower_bound(seg, m + 1);
  float s = 0.0f;
  for (int n = start; n < end; ++n) s += V[(size_t)n * DIM + e];
  out[(size_t)m * DIM + e] = s;
}

// ---------------------------------------------------------------------------
extern "C" void kernel_launch(void* const* d_in, const int* in_sizes, int n_in,
                              void* d_out, int out_size, void* d_ws, size_t ws_size,
                              hipStream_t stream) {
  const int*   fingerprints = (const int*)  d_in[0];
  const float* adjacencies  = (const float*)d_in[1];
  const int*   segment_ids  = (const int*)  d_in[2];
  const float* embed_table  = (const float*)d_in[3];
  const float* Wall         = (const float*)d_in[4];   // [3][64][64]
  const float* ball         = (const float*)d_in[5];   // [3][64]
  float* out = (float*)d_out;

  // workspace layout
  char* ws = (char*)d_ws;
  unsigned short* Abf = (unsigned short*)ws;                   // 128 MB
  ws += (size_t)N_ATOMS * N_ATOMS * sizeof(unsigned short);
  float* Vbuf = (float*)ws;                                    // 2 MB
  ws += (size_t)N_ATOMS * DIM * sizeof(float);
  float* Hbuf = (float*)ws;                                    // 2 MB
  ws += (size_t)N_ATOMS * DIM * sizeof(float);
  unsigned short* Ht = (unsigned short*)ws;                    // 1 MB
  ws += (size_t)DIM * N_ATOMS * sizeof(unsigned short);
  float* Part = (float*)ws;                                    // 8 MB

  // 1) adjacency f32 -> bf16 (L2-resident thereafter)
  {
    size_t nthreads = (size_t)N_ATOMS * N_ATOMS / 8;           // 8,388,608
    cvt_a_bf16<<<dim3((unsigned)(nthreads / 256)), dim3(256), 0, stream>>>(adjacencies, Abf);
  }
  // 2) embedding gather
  embed_gather<<<dim3(N_ATOMS * DIM / 4 / 256), dim3(256), 0, stream>>>(
      fingerprints, embed_table, Vbuf);

  // 3) message-passing layers
  for (int layer = 0; layer < N_LAYERS; ++layer) {
    linear_relu<<<dim3(N_ATOMS / 4), dim3(256), 0, stream>>>(
        Vbuf, Wall + (size_t)layer * DIM * DIM, ball + (size_t)layer * DIM,
        Hbuf, Ht);
    gemm_mp<<<dim3((N_ATOMS / 16) * KSPLIT / 4), dim3(128), 0, stream>>>(
        Abf, Ht, Part);
    combine<<<dim3(N_ATOMS * DIM / 4 / 256), dim3(256), 0, stream>>>(
        Hbuf, Part, Vbuf);
  }

  // 4) per-molecule sum pooling
  segment_sum<<<dim3(N_MOL), dim3(64), 0, stream>>>(Vbuf, segment_ids, out);
}
